// FPN_84035330114150
// MI455X (gfx1250) — compile-verified, matrix operand order checked by reading, not run on hardware
//
#include <hip/hip_runtime.h>
#include <hip/hip_bf16.h>
#include <math.h>

typedef __attribute__((ext_vector_type(16))) _Float16 v16h;
typedef __attribute__((ext_vector_type(8)))  _Float16 v8h;
typedef __attribute__((ext_vector_type(8)))  float    v8f;

union Frag16 { v16h v; v8h h[2]; };
union PackU  { unsigned int u; _Float16 f[2]; };

#define EPSV 1e-5f

__device__ __forceinline__ int iclampi(int v, int lo, int hi) {
  return v < lo ? lo : (v > hi ? hi : v);
}

// ---------------- f32 -> f16 weight conversion ----------------
__global__ void k_f32_to_f16(const float* __restrict__ src,
                             _Float16* __restrict__ dst, int n) {
  int i = blockIdx.x * blockDim.x + threadIdx.x;
  if (i < n) dst[i] = (_Float16)src[i];
}

// ---------------- WMMA GEMM: C[M,N] = A(f16)[M,K] * B[K,N] + bias ----------------
// B sources: materialized f16 cols stored TRANSPOSED [N,K], or implicit im2col
// (conv3x3, pad=1) of f32 X [Cin,H,W] with K = Cin*9, N = H*W.
// Block: 128 threads = 4 waves. Block tile: 64(M) x 64(N), K stepped by 32.
// Each wave holds one A fragment and issues 4 WMMAs (4 N-subtiles) per K step,
// using the reuse-A operand cache on the first three.
__global__ void __launch_bounds__(128)
k_gemm_wmma(const _Float16* __restrict__ A,
            const _Float16* __restrict__ BcolsT,   // [N,K]
            const float* __restrict__ X,
            const float* __restrict__ bias,
            float* __restrict__ C,
            int M, int K, int N, int H, int W,
            int implicit_conv, int relu) {
  __shared__ uint4 As4[64][4];   // 64 rows (m) x 32 halves (k)
  __shared__ uint4 Bs4[64][4];   // 64 cols (n) x 32 halves (k)  [transposed]
  const int t    = threadIdx.x;       // 0..127
  const int wave = t >> 5;            // 0..3
  const int lane = t & 31;
  const int lrow = lane & 15;
  const int hsel = lane >> 4;         // lane half selects K sub-range
  const int Nbase = blockIdx.x * 64;
  const int Mbase = blockIdx.y * 64;

  v8f acc0 = {0.f,0.f,0.f,0.f,0.f,0.f,0.f,0.f};
  v8f acc1 = acc0, acc2 = acc0, acc3 = acc0;

  // A staging: row = t>>1 (0..63), 32-byte chunk = t&1
  const int aml = t >> 1;
  const int ach = t & 1;
  const int agm = Mbase + aml;
  // B staging: each thread owns one column nl and 16 consecutive k halves
  const int nl    = t >> 1;           // 0..63
  const int kbase = (t & 1) * 16;     // 0 or 16
  const int bn    = Nbase + nl;
  const int bhh   = bn / W;
  const int bww   = bn - bhh * W;

  for (int kc = 0; kc < K; kc += 32) {
    // ---- stage A: 2x global b128 + 2x ds b128 per thread ----
    uint4 av0 = {0u,0u,0u,0u}, av1 = av0;
    if (agm < M) {
      const uint4* src = (const uint4*)(A + (size_t)agm * K + kc + ach * 16);
      av0 = src[0];
      av1 = src[1];
      if (kc + 32 < K)
        __builtin_prefetch(A + (size_t)agm * K + kc + 32 + ach * 16, 0, 3);
    }
    As4[aml][ach * 2]     = av0;
    As4[aml][ach * 2 + 1] = av1;

    // ---- stage B (transposed): 16 consecutive halves per thread ----
    if (implicit_conv) {
      // one divide per K-step, then pure increment tap walk
      int s = kc + kbase;
      unsigned cu = (unsigned)s / 9u;
      unsigned ru = (unsigned)s - cu * 9u;
      int ccc = (int)cu;
      int kyy = (int)(ru / 3u);
      int kxx = (int)(ru - (unsigned)kyy * 3u);
      PackU pk[8];
#pragma unroll
      for (int j = 0; j < 16; ++j) {
        int ih = bhh + kyy - 1, iw = bww + kxx - 1;
        float v = (ih >= 0 && ih < H && iw >= 0 && iw < W)
                      ? X[((size_t)ccc * H + ih) * W + iw] : 0.f;
        pk[j >> 1].f[j & 1] = (_Float16)v;
        if (++kxx == 3) { kxx = 0; if (++kyy == 3) { kyy = 0; ++ccc; } }
      }
      uint4 q0, q1;
      q0.x = pk[0].u; q0.y = pk[1].u; q0.z = pk[2].u; q0.w = pk[3].u;
      q1.x = pk[4].u; q1.y = pk[5].u; q1.z = pk[6].u; q1.w = pk[7].u;
      Bs4[nl][ach * 2]     = q0;
      Bs4[nl][ach * 2 + 1] = q1;
    } else {
      const uint4* src = (const uint4*)(BcolsT + (size_t)bn * K + kc + kbase);
      Bs4[nl][ach * 2]     = src[0];
      Bs4[nl][ach * 2 + 1] = src[1];
      if (kc + 32 < K)
        __builtin_prefetch(BcolsT + (size_t)bn * K + kc + 32 + kbase, 0, 3);
    }
    __syncthreads();

    // ---- fragments: 2x ds_load_b128 per operand ----
    Frag16 fa;
    fa.h[0] = *(const v8h*)&As4[wave * 16 + lrow][hsel];
    fa.h[1] = *(const v8h*)&As4[wave * 16 + lrow][2 + hsel];
    Frag16 fb0, fb1, fb2, fb3;
    fb0.h[0] = *(const v8h*)&Bs4[lrow][hsel];
    fb0.h[1] = *(const v8h*)&Bs4[lrow][2 + hsel];
    fb1.h[0] = *(const v8h*)&Bs4[16 + lrow][hsel];
    fb1.h[1] = *(const v8h*)&Bs4[16 + lrow][2 + hsel];
    fb2.h[0] = *(const v8h*)&Bs4[32 + lrow][hsel];
    fb2.h[1] = *(const v8h*)&Bs4[32 + lrow][2 + hsel];
    fb3.h[0] = *(const v8h*)&Bs4[48 + lrow][hsel];
    fb3.h[1] = *(const v8h*)&Bs4[48 + lrow][2 + hsel];

    // 4 identical back-to-back WMMAs sharing A -> reuse-A hint on first three
    acc0 = __builtin_amdgcn_wmma_f32_16x16x32_f16(false, fa.v, false, fb0.v,
                                                  (short)0, acc0, true,  false);
    acc1 = __builtin_amdgcn_wmma_f32_16x16x32_f16(false, fa.v, false, fb1.v,
                                                  (short)0, acc1, true,  false);
    acc2 = __builtin_amdgcn_wmma_f32_16x16x32_f16(false, fa.v, false, fb2.v,
                                                  (short)0, acc2, true,  false);
    acc3 = __builtin_amdgcn_wmma_f32_16x16x32_f16(false, fa.v, false, fb3.v,
                                                  (short)0, acc3, false, false);
    __syncthreads();
  }

  // ---- epilogue: C/D 16x16 f32 layout (VGPR r -> M=r / r+8 by lane half) ----
#pragma unroll
  for (int r = 0; r < 8; ++r) {
    int m = Mbase + wave * 16 + (hsel ? r + 8 : r);
    if (m < M) {
      float bv = bias ? bias[m] : 0.f;
      float* crow = C + (size_t)m * N + Nbase + lrow;
      float v0 = acc0[r] + bv, v1 = acc1[r] + bv;
      float v2 = acc2[r] + bv, v3 = acc3[r] + bv;
      if (relu) {
        v0 = fmaxf(v0, 0.f); v1 = fmaxf(v1, 0.f);
        v2 = fmaxf(v2, 0.f); v3 = fmaxf(v3, 0.f);
      }
      crow[0]  = v0;
      crow[16] = v1;
      crow[32] = v2;
      crow[48] = v3;
    }
  }
}

// ---------------- DCN v2 modulated bilinear sampling -> f16 colsT [N, Cin*9] ----------------
// flat output index i = (n*Cin + c)*9 + k  (contiguous per-thread writes)
__global__ void k_dcn_sample(const float* __restrict__ x,
                             const float* __restrict__ off,
                             _Float16* __restrict__ colsT,
                             int Cin, int H, int W, int groups) {
  const int HW = H * W;
  const size_t total = (size_t)Cin * 9 * HW;
  size_t i = (size_t)blockIdx.x * blockDim.x + threadIdx.x;
  if (i >= total) return;
  int k = (int)(i % 9);
  size_t r = i / 9;
  int c = (int)(r % Cin);
  int n = (int)(r / Cin);
  int Cg = Cin / groups;
  int g  = c / Cg;
  int gk = g * 9 + k;
  float dy = off[(size_t)gk * HW + n];
  float dx = off[((size_t)(groups * 9) + gk) * HW + n];
  float mv = off[((size_t)(2 * groups * 9) + gk) * HW + n];
  float mask = 1.f / (1.f + __expf(-mv));
  int hh = n / W, wwx = n - hh * W;
  float ys = (float)hh + (float)(k / 3 - 1) + dy;
  float xs = (float)wwx + (float)(k % 3 - 1) + dx;
  float fy = floorf(ys), fx = floorf(xs);
  float wy = ys - fy, wx = xs - fx;
  int y0 = (int)fy, x0 = (int)fx;
  const float* xc = x + (size_t)c * HW;
  auto samp = [&](int yy, int xx) -> float {
    if (yy < 0 || yy >= H || xx < 0 || xx >= W) return 0.f;
    return xc[yy * W + xx];
  };
  float v = (1.f - wy) * (1.f - wx) * samp(y0, x0)
          + (1.f - wy) * wx         * samp(y0, x0 + 1)
          + wy         * (1.f - wx) * samp(y0 + 1, x0)
          + wy         * wx         * samp(y0 + 1, x0 + 1);
  colsT[i] = (_Float16)(v * mask);
}

// ---------------- training-mode BatchNorm (B=1), optional affine + relu ----------------
__global__ void k_bn(const float* __restrict__ in, float* __restrict__ out,
                     const float* __restrict__ gamma, const float* __restrict__ beta,
                     int HW, int relu) {
  __shared__ float s1[256], s2[256];
  int c = blockIdx.x;
  const float* src = in + (size_t)c * HW;
  float sum = 0.f, sq = 0.f;
  for (int i = threadIdx.x; i < HW; i += 256) {
    float v = src[i]; sum += v; sq += v * v;
  }
  s1[threadIdx.x] = sum; s2[threadIdx.x] = sq;
  __syncthreads();
  for (int s = 128; s > 0; s >>= 1) {
    if ((int)threadIdx.x < s) {
      s1[threadIdx.x] += s1[threadIdx.x + s];
      s2[threadIdx.x] += s2[threadIdx.x + s];
    }
    __syncthreads();
  }
  float mu  = s1[0] / HW;
  float var = s2[0] / HW - mu * mu;
  float inv = rsqrtf(var + EPSV);
  float g = gamma ? gamma[c] : 1.f;
  float b = beta ? beta[c] : 0.f;
  float* dst = out + (size_t)c * HW;
  for (int i = threadIdx.x; i < HW; i += 256) {
    float v = (src[i] - mu) * inv * g + b;
    if (relu) v = fmaxf(v, 0.f);
    dst[i] = v;
  }
}

// ---------------- global average pool per channel ----------------
__global__ void k_gap(const float* __restrict__ in, float* __restrict__ a, int HW) {
  __shared__ float s1[256];
  int c = blockIdx.x;
  float sum = 0.f;
  for (int i = threadIdx.x; i < HW; i += 256) sum += in[(size_t)c * HW + i];
  s1[threadIdx.x] = sum;
  __syncthreads();
  for (int s = 128; s > 0; s >>= 1) {
    if ((int)threadIdx.x < s) s1[threadIdx.x] += s1[threadIdx.x + s];
    __syncthreads();
  }
  if (threadIdx.x == 0) a[c] = s1[0] / HW;
}

// ---------------- SE: relu(a @ w1^T) -> sigmoid(h @ w2^T). w1:[32,128] w2:[128,32] ----------------
__global__ void k_se_fc(const float* __restrict__ a, const float* __restrict__ w1,
                        const float* __restrict__ w2, float* __restrict__ gamma) {
  __shared__ float hbuf[32];
  __shared__ float av[128];
  int t = threadIdx.x;  // 128 threads
  av[t] = a[t];
  __syncthreads();
  if (t < 32) {
    float s = 0.f;
    for (int c = 0; c < 128; ++c) s += av[c] * w1[t * 128 + c];
    hbuf[t] = fmaxf(s, 0.f);
  }
  __syncthreads();
  float s = 0.f;
  for (int j = 0; j < 32; ++j) s += hbuf[j] * w2[t * 32 + j];
  gamma[t] = 1.f / (1.f + __expf(-s));
}

// ---------------- p = relu(bn*(1+gamma) + beta) ----------------
__global__ void k_combine(const float* __restrict__ bn, const float* __restrict__ gamma,
                          const float* __restrict__ beta, float* __restrict__ out, int HW) {
  int i = blockIdx.x * blockDim.x + threadIdx.x;
  if (i < 128 * HW) {
    int c = i / HW;
    out[i] = fmaxf(bn[i] * (1.f + gamma[c]) + beta[i], 0.f);
  }
}

// ---------------- half-pixel bilinear 2x upsample ----------------
__global__ void k_upsample2x(const float* __restrict__ in, float* __restrict__ out,
                             int C, int H, int W) {
  int Ho = H * 2, Wo = W * 2;
  size_t total = (size_t)C * Ho * Wo;
  size_t i = (size_t)blockIdx.x * blockDim.x + threadIdx.x;
  if (i >= total) return;
  int xo = (int)(i % Wo);
  int yo = (int)((i / Wo) % Ho);
  int c  = (int)(i / ((size_t)Wo * Ho));
  float sy = (yo + 0.5f) * 0.5f - 0.5f;
  float sx = (xo + 0.5f) * 0.5f - 0.5f;
  float fy = floorf(sy), fx = floorf(sx);
  float ty = sy - fy, tx = sx - fx;
  int y0 = (int)fy, x0 = (int)fx;
  int y0c = iclampi(y0, 0, H - 1), y1c = iclampi(y0 + 1, 0, H - 1);
  int x0c = iclampi(x0, 0, W - 1), x1c = iclampi(x0 + 1, 0, W - 1);
  const float* src = in + (size_t)c * H * W;
  float v = (1.f - ty) * (1.f - tx) * src[y0c * W + x0c]
          + (1.f - ty) * tx         * src[y0c * W + x1c]
          + ty         * (1.f - tx) * src[y1c * W + x0c]
          + ty         * tx         * src[y1c * W + x1c];
  out[i] = v;
}

extern "C" void kernel_launch(void* const* d_in, const int* in_sizes, int n_in,
                              void* d_out, int out_size, void* d_ws, size_t ws_size,
                              hipStream_t stream) {
  (void)in_sizes; (void)n_in; (void)out_size; (void)ws_size;
  const size_t MB = 1024 * 1024;
  char* ws = (char*)d_ws;
  _Float16* wg16   = (_Float16*)(ws + 0);        //  8 MB: f16 weight scratch
  _Float16* colsT  = (_Float16*)(ws + 8 * MB);   // 80 MB: deformable cols (f16, [N,K])
  float*    offb   = (float*)(ws + 88 * MB);     //  8 MB: offset conv output
  float*    yb     = (float*)(ws + 96 * MB);     //  9 MB: DCN GEMM output
  float*    bnb    = (float*)(ws + 105 * MB);    //  9 MB: BN output
  float*    upb    = (float*)(ws + 114 * MB);    //  9 MB: upsampled prev-level p
  float*    t1b    = (float*)(ws + 123 * MB);    //  9 MB: beta conv1 output
  float*    betab  = (float*)(ws + 132 * MB);    //  9 MB: beta output
  float*    av     = (float*)(ws + 141 * MB);    // small: GAP vector
  float*    gammav = av + 256;                   // small: SE gamma vector
  float*    pb     = (float*)(ws + 142 * MB);    //  9 MB: current level p

  auto cvt = [&](const float* s, _Float16* d, int n) {
    k_f32_to_f16<<<(n + 255) / 256, 256, 0, stream>>>(s, d, n);
  };
  auto gemm = [&](const _Float16* A, const _Float16* BT, const float* X,
                  const float* bias, float* Cd, int M, int K, int N,
                  int H, int W, int impl, int relu) {
    dim3 grid(N / 64, (M + 63) / 64);
    k_gemm_wmma<<<grid, 128, 0, stream>>>(A, BT, X, bias, Cd, M, K, N, H, W, impl, relu);
  };

  // levels processed top-down: 5, 4, 3, 2
  const int   xin[4]    = {3, 2, 1, 0};
  const int   Cin_[4]   = {2048, 1024, 512, 256};
  const int   Hh_[4]    = {16, 32, 64, 128};
  const int   wi_[4]    = {22, 16, 10, 4};     // p{l}_w, _b, _wo, _bo, _sm_w, _sm_b
  const int   gi_[4]    = {-1, 42, 36, 30};    // g_w1, g_w2, be_w1, be_b1, be_w2, be_b2
  const size_t ooff_[4] = {2752512, 2621440, 2097152, 0}; // p5o,p4o,p3o,p2o in d_out

  for (int li = 0; li < 4; ++li) {
    const float* x   = (const float*)d_in[xin[li]];
    const int Cin    = Cin_[li];
    const int H      = Hh_[li];
    const int W      = H;
    const int HW     = H * W;
    const int K9     = Cin * 9;
    const int wi     = wi_[li];
    const int gi     = gi_[li];
    const float* p_w  = (const float*)d_in[wi + 0];
    const float* p_b  = (const float*)d_in[wi + 1];
    const float* p_wo = (const float*)d_in[wi + 2];
    const float* p_bo = (const float*)d_in[wi + 3];
    const float* sm_w = (const float*)d_in[wi + 4];
    const float* sm_b = (const float*)d_in[wi + 5];

    // 1) offset conv (3*G*9 = 108 channels) via implicit-im2col WMMA GEMM
    cvt(p_wo, wg16, 108 * K9);
    gemm(wg16, nullptr, x, p_bo, offb, 108, K9, HW, H, W, 1, 0);

    // 2) modulated deformable sampling -> f16 colsT [HW, Cin*9]
    {
      size_t tot = (size_t)K9 * HW;
      k_dcn_sample<<<(unsigned)((tot + 255) / 256), 256, 0, stream>>>(
          x, offb, colsT, Cin, H, W, 4);
    }

    // 3) main DCN GEMM: y = W[128, Cin*9] x cols + b
    cvt(p_w, wg16, 128 * K9);
    gemm(wg16, colsT, nullptr, p_b, yb, 128, K9, HW, H, W, 0, 0);

    if (li == 0) {
      // p5 = relu(BN(y, gamma, beta))
      k_bn<<<128, 256, 0, stream>>>(yb, pb, (const float*)d_in[28],
                                    (const float*)d_in[29], HW, 1);
    } else {
      // BN without affine; SE gamma + beta path from upsampled previous level
      k_bn<<<128, 256, 0, stream>>>(yb, bnb, nullptr, nullptr, HW, 0);
      k_gap<<<128, 256, 0, stream>>>(upb, av, HW);
      k_se_fc<<<1, 128, 0, stream>>>(av, (const float*)d_in[gi + 0],
                                     (const float*)d_in[gi + 1], gammav);
      cvt((const float*)d_in[gi + 2], wg16, 128 * 128 * 9);
      gemm(wg16, nullptr, upb, (const float*)d_in[gi + 3], t1b,
           128, 1152, HW, H, W, 1, 1);
      cvt((const float*)d_in[gi + 4], wg16, 128 * 128 * 9);
      gemm(wg16, nullptr, t1b, (const float*)d_in[gi + 5], betab,
           128, 1152, HW, H, W, 1, 0);
      k_combine<<<(128 * HW + 255) / 256, 256, 0, stream>>>(bnb, gammav, betab, pb, HW);
    }

    // 4) smoothing conv3x3 -> straight into d_out slice
    cvt(sm_w, wg16, 128 * 128 * 9);
    gemm(wg16, nullptr, pb, sm_b, (float*)d_out + ooff_[li],
         128, 1152, HW, H, W, 1, 0);

    // 5) upsample p for next (finer) level
    if (li < 3) {
      size_t tot = (size_t)128 * HW * 4;
      k_upsample2x<<<(unsigned)((tot + 255) / 256), 256, 0, stream>>>(pb, upb, 128, H, W);
    }
  }
}
